// LowerBlock_73486890434738
// MI455X (gfx1250) — matrix operand order repaired and checked by reading, not compiled, not run on hardware
//
#include <hip/hip_runtime.h>

// ---------------- problem constants (match reference) ----------------
constexpr int B_   = 2;
constexpr int S_   = 2048;
constexpr int D_   = 1024;
constexpr int H_   = 16;
constexpr int W_   = 16;     // sliding window
constexpr int HD_  = 64;     // head dim
constexpr int DFF_ = 4096;
constexpr int M_   = B_ * S_;   // 4096 tokens

// ---------------- WMMA types ----------------
typedef __bf16        v16bf __attribute__((ext_vector_type(16)));
typedef float         v8f   __attribute__((ext_vector_type(8)));
typedef unsigned int  u32x4 __attribute__((ext_vector_type(4)));

union Frag { v16bf v; u32x4 u[2]; };

__device__ __forceinline__ void bf16_split(float x, __bf16& hi, __bf16& lo) {
    hi = (__bf16)x;
    lo = (__bf16)(x - (float)hi);
}

// error-compensated bf16x3 accumulate: c += Al*Bh + Ah*Bl + Ah*Bh
__device__ __forceinline__ v8f wmma3(const Frag& ah, const Frag& al,
                                     const Frag& bh, const Frag& bl, v8f c) {
    c = __builtin_amdgcn_wmma_f32_16x16x32_bf16(false, al.v, false, bh.v,
                                                (short)0, c, false, false);
    c = __builtin_amdgcn_wmma_f32_16x16x32_bf16(false, ah.v, false, bl.v,
                                                (short)0, c, false, false);
    c = __builtin_amdgcn_wmma_f32_16x16x32_bf16(false, ah.v, false, bh.v,
                                                (short)0, c, false, false);
    return c;
}

// ---------------- f32 -> bf16 hi/lo split (weights) ----------------
__global__ __launch_bounds__(256) void split_kernel(const float* __restrict__ w,
                                                    __bf16* __restrict__ hi,
                                                    __bf16* __restrict__ lo,
                                                    size_t n) {
    for (size_t i = (size_t)blockIdx.x * 256 + threadIdx.x; i < n;
         i += (size_t)gridDim.x * 256) {
        bf16_split(w[i], hi[i], lo[i]);
    }
}

// ---------------- LayerNorm + bf16 split (one block per token row) ----------------
__global__ __launch_bounds__(256) void ln_split_kernel(const float* __restrict__ x,
                                                       const float* __restrict__ g,
                                                       const float* __restrict__ b,
                                                       __bf16* __restrict__ hi,
                                                       __bf16* __restrict__ lo) {
    const int row = blockIdx.x;
    const int tid = threadIdx.x;
    const float* xr = x + (size_t)row * D_;
    __shared__ float ssum[256];
    __shared__ float ssq[256];
    float s = 0.f, s2 = 0.f;
    for (int i = tid; i < D_; i += 256) { float v = xr[i]; s += v; s2 += v * v; }
    ssum[tid] = s; ssq[tid] = s2;
    __syncthreads();
    for (int st = 128; st > 0; st >>= 1) {
        if (tid < st) { ssum[tid] += ssum[tid + st]; ssq[tid] += ssq[tid + st]; }
        __syncthreads();
    }
    const float mean = ssum[0] * (1.f / D_);
    const float var  = ssq[0] * (1.f / D_) - mean * mean;
    const float rinv = rsqrtf(var + 1e-5f);
    for (int i = tid; i < D_; i += 256) {
        float y = (xr[i] - mean) * rinv * g[i] + b[i];
        size_t idx = (size_t)row * D_ + i;
        bf16_split(y, hi[idx], lo[idx]);
    }
}

// ---------------- bf16x3 WMMA GEMM:  out[M,N] = A[M,K] * W[N,K]^T + bias (+epilogue)
// MODE 0: outF = c + bias
// MODE 1: outF = c + bias + res
// MODE 2: gelu(c + bias) -> bf16 hi/lo
// Block = 256 threads = 8 waves (2M x 4N); each wave owns a 2x2 grid of 16x16
// tiles (32M x 32N) -> block covers 64M x 128N. Per K-step: 16 b128 loads feed
// 12 WMMAs (2x the arithmetic intensity of one-tile-per-wave).
template <int MODE>
__global__ __launch_bounds__(256) void gemm_bf16x3_kernel(
    const __bf16* __restrict__ Ahi, const __bf16* __restrict__ Alo,
    const __bf16* __restrict__ Whi, const __bf16* __restrict__ Wlo,
    const float* __restrict__ bias, const float* __restrict__ res,
    float* __restrict__ outF, __bf16* __restrict__ outHi, __bf16* __restrict__ outLo,
    int N, int K) {
    const int lane = threadIdx.x & 31;
    const int wave = threadIdx.x >> 5;
    const int wm = wave >> 2;       // 0..1
    const int wn = wave & 3;        // 0..3
    const int mb = blockIdx.y * 64 + wm * 32;
    const int nb = blockIdx.x * 128 + wn * 32;
    const int half = lane >> 4;     // 0: lanes 0-15, 1: lanes 16-31
    const int r    = lane & 15;     // A row within tile / W row (= C column)

    const size_t aRow0 = (size_t)(mb + r) * K;
    const size_t aRow1 = (size_t)(mb + 16 + r) * K;
    const size_t bRow0 = (size_t)(nb + r) * K;
    const size_t bRow1 = (size_t)(nb + 16 + r) * K;

    v8f acc[2][2];
#pragma unroll
    for (int mi = 0; mi < 2; ++mi)
#pragma unroll
        for (int ni = 0; ni < 2; ++ni)
            acc[mi][ni] = (v8f){0.f, 0.f, 0.f, 0.f, 0.f, 0.f, 0.f, 0.f};

    for (int k0 = 0; k0 < K; k0 += 32) {
        Frag ah0, al0, ah1, al1, bh0, bl0, bh1, bl1;
        const int ka = k0 + half * 8;   // A: elems 0..7 = K ka.. ; 8..15 = K ka+16..
        ah0.u[0] = *(const u32x4*)(Ahi + aRow0 + ka);
        ah0.u[1] = *(const u32x4*)(Ahi + aRow0 + ka + 16);
        al0.u[0] = *(const u32x4*)(Alo + aRow0 + ka);
        al0.u[1] = *(const u32x4*)(Alo + aRow0 + ka + 16);
        ah1.u[0] = *(const u32x4*)(Ahi + aRow1 + ka);
        ah1.u[1] = *(const u32x4*)(Ahi + aRow1 + ka + 16);
        al1.u[0] = *(const u32x4*)(Alo + aRow1 + ka);
        al1.u[1] = *(const u32x4*)(Alo + aRow1 + ka + 16);
        const int kb = k0 + half * 8;   // B: elems 0..7 = K kb.. ; 8..15 = K kb+16..
        bh0.u[0] = *(const u32x4*)(Whi + bRow0 + kb);
        bh0.u[1] = *(const u32x4*)(Whi + bRow0 + kb + 16);
        bl0.u[0] = *(const u32x4*)(Wlo + bRow0 + kb);
        bl0.u[1] = *(const u32x4*)(Wlo + bRow0 + kb + 16);
        bh1.u[0] = *(const u32x4*)(Whi + bRow1 + kb);
        bh1.u[1] = *(const u32x4*)(Whi + bRow1 + kb + 16);
        bl1.u[0] = *(const u32x4*)(Wlo + bRow1 + kb);
        bl1.u[1] = *(const u32x4*)(Wlo + bRow1 + kb + 16);

        acc[0][0] = wmma3(ah0, al0, bh0, bl0, acc[0][0]);
        acc[0][1] = wmma3(ah0, al0, bh1, bl1, acc[0][1]);
        acc[1][0] = wmma3(ah1, al1, bh0, bl0, acc[1][0]);
        acc[1][1] = wmma3(ah1, al1, bh1, bl1, acc[1][1]);
    }

#pragma unroll
    for (int mi = 0; mi < 2; ++mi) {
#pragma unroll
        for (int ni = 0; ni < 2; ++ni) {
            const int   cn    = nb + ni * 16 + r;        // C column
            const int   mBase = mb + mi * 16 + half * 8; // C rows mBase..mBase+7
            const float bn    = bias[cn];
#pragma unroll
            for (int j = 0; j < 8; ++j) {
                float c = acc[mi][ni][j] + bn;
                size_t idx = (size_t)(mBase + j) * N + cn;
                if (MODE == 1) c += res[idx];
                if (MODE == 2) {
                    float t  = 0.7978845608028654f * (c + 0.044715f * c * c * c);
                    float gl = 0.5f * c * (1.0f + tanhf(t));
                    bf16_split(gl, outHi[idx], outLo[idx]);
                } else {
                    outF[idx] = c;
                }
            }
        }
    }
}

// ---------------- sliding-window attention: softmax probs ----------------
// one thread per (b, h, q); p[b][h][q][j] with key index q-(W-1)+j
__global__ __launch_bounds__(256) void attn_probs_kernel(const float* __restrict__ qkv,
                                                         float* __restrict__ p) {
    const int t = blockIdx.x * 256 + threadIdx.x;   // 0 .. B*H*S-1
    const int q = t % S_;
    const int h = (t / S_) % H_;
    const int b = t / (S_ * H_);
    const float* qr = qkv + (size_t)(b * S_ + q) * (3 * D_) + h * HD_;
    float sc[W_];
    float mx = -3.0e38f;
#pragma unroll
    for (int j = 0; j < W_; ++j) {
        const int kidx = q - (W_ - 1) + j;
        if (kidx < 0) { sc[j] = -3.0e38f; continue; }
        const float* kr = qkv + (size_t)(b * S_ + kidx) * (3 * D_) + D_ + h * HD_;
        float acc = 0.f;
        for (int d = 0; d < HD_; ++d) acc += qr[d] * kr[d];
        sc[j] = acc * 0.125f;   // 1/sqrt(64)
        mx = fmaxf(mx, sc[j]);
    }
    float den = 0.f;
#pragma unroll
    for (int j = 0; j < W_; ++j) {
        float e = (sc[j] > -1.0e38f) ? __expf(sc[j] - mx) : 0.f;
        sc[j] = e;
        den += e;
    }
    const float inv = 1.f / den;
    float* pr = p + (size_t)((b * H_ + h) * S_ + q) * W_;
#pragma unroll
    for (int j = 0; j < W_; ++j) pr[j] = sc[j] * inv;
}

// ---------------- attention output: o = P @ V, stored as bf16 hi/lo ----------------
// one thread per (b, q, channel)
__global__ __launch_bounds__(256) void attn_out_kernel(const float* __restrict__ qkv,
                                                       const float* __restrict__ p,
                                                       __bf16* __restrict__ ohi,
                                                       __bf16* __restrict__ olo) {
    const size_t t = (size_t)blockIdx.x * 256 + threadIdx.x;  // 0 .. M*D-1
    const int dcol = (int)(t % D_);
    const int q    = (int)((t / D_) % S_);
    const int b    = (int)(t / ((size_t)D_ * S_));
    const int h    = dcol / HD_;
    const float* pr = p + (size_t)((b * H_ + h) * S_ + q) * W_;
    float acc = 0.f;
#pragma unroll
    for (int j = 0; j < W_; ++j) {
        const int kidx = q - (W_ - 1) + j;
        if (kidx >= 0)
            acc += pr[j] * qkv[(size_t)(b * S_ + kidx) * (3 * D_) + 2 * D_ + dcol];
    }
    bf16_split(acc, ohi[t], olo[t]);
}

// ---------------- host driver ----------------
extern "C" void kernel_launch(void* const* d_in, const int* in_sizes, int n_in,
                              void* d_out, int out_size, void* d_ws, size_t ws_size,
                              hipStream_t stream) {
    (void)in_sizes; (void)n_in; (void)out_size; (void)ws_size;

    const float* x          = (const float*)d_in[0];
    const float* ln1_g      = (const float*)d_in[1];
    const float* ln1_b      = (const float*)d_in[2];
    const float* in_proj_w  = (const float*)d_in[3];
    const float* in_proj_b  = (const float*)d_in[4];
    const float* out_proj_w = (const float*)d_in[5];
    const float* out_proj_b = (const float*)d_in[6];
    const float* ln2_g      = (const float*)d_in[7];
    const float* ln2_b      = (const float*)d_in[8];
    const float* mlp_w1     = (const float*)d_in[9];
    const float* mlp_b1     = (const float*)d_in[10];
    const float* mlp_w2     = (const float*)d_in[11];
    const float* mlp_b2     = (const float*)d_in[12];

    char* ws = (char*)d_ws;
    size_t off = 0;
    auto alloc = [&](size_t bytes) -> char* {
        char* pp = ws + off;
        off += (bytes + 255) & ~(size_t)255;
        return pp;
    };

    const size_t WQ_E = (size_t)3 * D_ * D_;
    const size_t WO_E = (size_t)D_ * D_;
    const size_t W1_E = (size_t)DFF_ * D_;
    const size_t W2_E = (size_t)D_ * DFF_;
    const size_t HM_E = (size_t)M_ * D_;

    // persistent region
    __bf16* wq_hi = (__bf16*)alloc(WQ_E * 2);
    __bf16* wq_lo = (__bf16*)alloc(WQ_E * 2);
    __bf16* wo_hi = (__bf16*)alloc(WO_E * 2);
    __bf16* wo_lo = (__bf16*)alloc(WO_E * 2);
    __bf16* w1_hi = (__bf16*)alloc(W1_E * 2);
    __bf16* w1_lo = (__bf16*)alloc(W1_E * 2);
    __bf16* w2_hi = (__bf16*)alloc(W2_E * 2);
    __bf16* w2_lo = (__bf16*)alloc(W2_E * 2);
    __bf16* h_hi  = (__bf16*)alloc(HM_E * 2);
    __bf16* h_lo  = (__bf16*)alloc(HM_E * 2);
    float*  x1    = (float*) alloc(HM_E * 4);
    __bf16* xn_hi = (__bf16*)alloc(HM_E * 2);
    __bf16* xn_lo = (__bf16*)alloc(HM_E * 2);

    // transient region (aliased): {qkv, probs, o} overlapped later by {hidden}
    const size_t QKV_BYTES = (size_t)M_ * 3 * D_ * 4;          // 48 MB
    const size_t P_BYTES   = (size_t)B_ * H_ * S_ * W_ * 4;    // 4 MB
    const size_t O_BYTES   = HM_E * 2;                         // 8 MB each
    const size_t HID_BYTES = (size_t)M_ * DFF_ * 2;            // 32 MB each
    char* tbase = ws + off;
    float*  qkv    = (float*) (tbase);
    float*  probs  = (float*) (tbase + QKV_BYTES);
    __bf16* o_hi   = (__bf16*)(tbase + QKV_BYTES + P_BYTES);
    __bf16* o_lo   = (__bf16*)(tbase + QKV_BYTES + P_BYTES + O_BYTES);
    __bf16* hid_hi = (__bf16*)(tbase);
    __bf16* hid_lo = (__bf16*)(tbase + HID_BYTES);

    // 1. weight splits
    split_kernel<<<1024, 256, 0, stream>>>(in_proj_w,  wq_hi, wq_lo, WQ_E);
    split_kernel<<<512,  256, 0, stream>>>(out_proj_w, wo_hi, wo_lo, WO_E);
    split_kernel<<<1024, 256, 0, stream>>>(mlp_w1,     w1_hi, w1_lo, W1_E);
    split_kernel<<<1024, 256, 0, stream>>>(mlp_w2,     w2_hi, w2_lo, W2_E);

    // 2. LN1
    ln_split_kernel<<<M_, 256, 0, stream>>>(x, ln1_g, ln1_b, h_hi, h_lo);

    // 3. QKV = h @ in_proj_w^T + b      [4096 x 3072]
    gemm_bf16x3_kernel<0><<<dim3((3 * D_) / 128, M_ / 64), 256, 0, stream>>>(
        h_hi, h_lo, wq_hi, wq_lo, in_proj_b, nullptr, qkv, nullptr, nullptr,
        3 * D_, D_);

    // 4-5. sliding-window attention
    attn_probs_kernel<<<(B_ * H_ * S_) / 256, 256, 0, stream>>>(qkv, probs);
    attn_out_kernel<<<(unsigned)((size_t)M_ * D_ / 256), 256, 0, stream>>>(
        qkv, probs, o_hi, o_lo);

    // 6. x1 = o @ out_proj_w^T + b + x  [4096 x 1024]
    gemm_bf16x3_kernel<1><<<dim3(D_ / 128, M_ / 64), 256, 0, stream>>>(
        o_hi, o_lo, wo_hi, wo_lo, out_proj_b, x, x1, nullptr, nullptr, D_, D_);

    // 7. LN2
    ln_split_kernel<<<M_, 256, 0, stream>>>(x1, ln2_g, ln2_b, xn_hi, xn_lo);

    // 8. hidden = gelu(xn @ w1^T + b1)  [4096 x 4096] -> bf16 hi/lo
    gemm_bf16x3_kernel<2><<<dim3(DFF_ / 128, M_ / 64), 256, 0, stream>>>(
        xn_hi, xn_lo, w1_hi, w1_lo, mlp_b1, nullptr, nullptr, hid_hi, hid_lo,
        DFF_, D_);

    // 9. out = hidden @ w2^T + b2 + x1  [4096 x 1024]
    gemm_bf16x3_kernel<1><<<dim3(D_ / 128, M_ / 64), 256, 0, stream>>>(
        hid_hi, hid_lo, w2_hi, w2_lo, mlp_b2, x1, (float*)d_out, nullptr,
        nullptr, D_, DFF_);
}
